// GNN_65841848648310
// MI455X (gfx1250) — matrix operand order repaired and checked by reading, not compiled, not run on hardware
//
#include <hip/hip_runtime.h>
#include <hip/hip_bf16.h>

// ---------------------------------------------------------------------------
// GraphConv GNN, 4 layers, N=50000, D=128, E=600000.
// Memory-bound: edge gather/scatter dominates; X and agg fit in 192MB L2.
// GEMMs done with v_wmma_f32_16x16x32_bf16 (bf16 in, f32 acc).
// 64-row M-tiles: B (weight) fragments live in registers and are reused
// across 4 M-subtiles, cutting weight re-reads from L2 by 4x.
// ---------------------------------------------------------------------------

typedef __attribute__((ext_vector_type(16))) __bf16 v16bf;
typedef __attribute__((ext_vector_type(8)))  float  v8f;

#define DIM   128
#define MROWS 64     // rows per block (4 subtiles of 16)

// ---- zero a float buffer (float4 vectorized) ------------------------------
__global__ void gnn_zero_f4(float4* p, int n4) {
    int i = blockIdx.x * blockDim.x + threadIdx.x;
    if (i < n4) p[i] = make_float4(0.f, 0.f, 0.f, 0.f);
}

// ---- degree accumulation ---------------------------------------------------
__global__ void gnn_degree(const int* __restrict__ dst, float* __restrict__ deg, int E) {
    int e = blockIdx.x * blockDim.x + threadIdx.x;
    if (e < E) atomicAdd(&deg[dst[e]], 1.0f);
}

__global__ void gnn_invdeg(float* deg, int N) {
    int i = blockIdx.x * blockDim.x + threadIdx.x;
    if (i < N) deg[i] = 1.0f / fmaxf(deg[i], 1.0f);
}

// ---- edge scatter: agg[dst] += X[src] -------------------------------------
// One wave (32 lanes) per edge; each lane handles 4 consecutive features.
// Uniform-address index loads broadcast within the wave; X rows are 512B
// contiguous and L2-resident, atomics resolve at L2.
__global__ __launch_bounds__(256)
void gnn_scatter(const float* __restrict__ X,
                 const int* __restrict__ src, const int* __restrict__ dst,
                 float* __restrict__ agg, int E) {
    int t    = blockIdx.x * blockDim.x + threadIdx.x;
    int e    = t >> 5;
    if (e >= E) return;
    int lane = t & 31;
    int s = src[e];
    int d = dst[e];
    const float4 x = ((const float4*)(X + (size_t)s * DIM))[lane];
    float* out = agg + (size_t)d * DIM + lane * 4;
    atomicAdd(out + 0, x.x);
    atomicAdd(out + 1, x.y);
    atomicAdd(out + 2, x.z);
    atomicAdd(out + 3, x.w);
}

// ---- fused (agg*invdeg)@Wrel + brel + X@Wroot, then ELU --------------------
// Block = 256 threads = 8 wave32. Block owns 64 node rows; wave w owns
// output columns [16w, 16w+16) and sweeps 4 row-subtiles, reusing its
// register-resident weight fragments. Xin may alias Xout (rows staged to
// LDS before any store).
__global__ __launch_bounds__(256)
void gnn_gemm_elu(const float* Xin,
                  const float* __restrict__ agg,
                  const float* __restrict__ invdeg,
                  const float* __restrict__ Wrel,
                  const float* __restrict__ brel,
                  const float* __restrict__ Wroot,
                  float* Xout, int N) {
    __shared__ __bf16 sX[MROWS][DIM + 8];   // +8 bf16 pad to spread LDS banks
    __shared__ __bf16 sA[MROWS][DIM + 8];

    const int tid = threadIdx.x;
    const int m0  = blockIdx.x * MROWS;

    // Cooperative stage of 64 rows of X and scaled agg into LDS as bf16.
    // float4-coalesced: 64*32 = 2048 float4 slots over 256 threads.
    for (int i = tid; i < MROWS * (DIM / 4); i += 256) {
        int r   = i >> 5;            // row within tile
        int c4  = (i & 31) * 4;      // starting column
        int row = m0 + r;
        if (row < N) {
            float  s  = invdeg[row];
            float4 xv = *(const float4*)(Xin + (size_t)row * DIM + c4);
            float4 av = *(const float4*)(agg + (size_t)row * DIM + c4);
            sX[r][c4 + 0] = (__bf16)xv.x; sX[r][c4 + 1] = (__bf16)xv.y;
            sX[r][c4 + 2] = (__bf16)xv.z; sX[r][c4 + 3] = (__bf16)xv.w;
            sA[r][c4 + 0] = (__bf16)(av.x * s); sA[r][c4 + 1] = (__bf16)(av.y * s);
            sA[r][c4 + 2] = (__bf16)(av.z * s); sA[r][c4 + 3] = (__bf16)(av.w * s);
        } else {
            sX[r][c4 + 0] = (__bf16)0.f; sX[r][c4 + 1] = (__bf16)0.f;
            sX[r][c4 + 2] = (__bf16)0.f; sX[r][c4 + 3] = (__bf16)0.f;
            sA[r][c4 + 0] = (__bf16)0.f; sA[r][c4 + 1] = (__bf16)0.f;
            sA[r][c4 + 2] = (__bf16)0.f; sA[r][c4 + 3] = (__bf16)0.f;
        }
    }
    __syncthreads();

    const int wave = tid >> 5;          // 0..7 -> output column tile
    const int lane = tid & 31;
    const int half = lane >> 4;         // lane-half per ISA fragment layout
    const int l15  = lane & 15;
    const int n0   = wave * 16;
    const int ncol = n0 + l15;

    // Preload B fragments (bf16 32x16): element e of chunk c holds
    // K = c*32 + 16*half + e, N = lane&15.  Reused across all 4 M-subtiles.
    v16bf brelf[4], brotf[4];
#pragma unroll
    for (int c = 0; c < 4; ++c) {
#pragma unroll
        for (int e = 0; e < 16; ++e) {
            int k = c * 32 + 16 * half + e;
            brelf[c][e] = (__bf16)Wrel [k * DIM + ncol];
            brotf[c][e] = (__bf16)Wroot[k * DIM + ncol];
        }
    }

    const float bias = brel[ncol];

#pragma unroll
    for (int mt = 0; mt < 4; ++mt) {
        const int rbase = mt * 16;
        v8f acc = {};
#pragma unroll
        for (int c = 0; c < 4; ++c) {
            const int kc = c * 32;
            // A fragment (bf16 16x32): lane row m=lane&15; element e maps to
            // K = 16*half + e (matches B fragment order).
            v16bf aAgg, aX;
#pragma unroll
            for (int e = 0; e < 16; ++e) {
                int k = 16 * half + e;
                aAgg[e] = sA[rbase + l15][kc + k];
                aX[e]   = sX[rbase + l15][kc + k];
            }
            acc = __builtin_amdgcn_wmma_f32_16x16x32_bf16(
                      false, aAgg, false, brelf[c], (short)0, acc, false, false);
            acc = __builtin_amdgcn_wmma_f32_16x16x32_bf16(
                      false, aX,   false, brotf[c], (short)0, acc, false, false);
        }

        // Bias (lin_rel only) + ELU + store. C/D layout: VGPR r -> M = r + 8*half.
#pragma unroll
        for (int r = 0; r < 8; ++r) {
            int row = m0 + rbase + r + 8 * half;
            if (row < N) {
                float v = acc[r] + bias;
                v = (v > 0.0f) ? v : (__expf(v) - 1.0f);
                Xout[(size_t)row * DIM + ncol] = v;
            }
        }
    }
}

// ---------------------------------------------------------------------------
extern "C" void kernel_launch(void* const* d_in, const int* in_sizes, int n_in,
                              void* d_out, int out_size, void* d_ws, size_t ws_size,
                              hipStream_t stream) {
    const float* node_emb = (const float*)d_in[0];   // [N, D]
    const int*   edge_idx = (const int*)  d_in[1];   // [2, E] flat: src then dst
    const float* Ws_rel   = (const float*)d_in[2];   // [L, D, D]
    const float* bs_rel   = (const float*)d_in[3];   // [L, D]
    const float* Ws_root  = (const float*)d_in[4];   // [L, D, D]
    float*       out      = (float*)d_out;           // [N, D]

    const int N = in_sizes[0] / DIM;
    const int E = in_sizes[1] / 2;
    const int L = in_sizes[3] / DIM;

    const int* src = edge_idx;
    const int* dst = edge_idx + E;

    // Workspace: deg/invdeg (N f32) then agg (N*D f32) ~ 25.8 MB.
    float* deg = (float*)d_ws;
    float* agg = deg + N;

    const int TB = 256;

    // Degree -> inverse degree (recomputed every call; no cached state).
    gnn_zero_f4<<<((N + 3) / 4 + TB - 1) / TB, TB, 0, stream>>>((float4*)deg, (N + 3) / 4);
    gnn_degree <<<(E + TB - 1) / TB, TB, 0, stream>>>(dst, deg, E);
    gnn_invdeg <<<(N + TB - 1) / TB, TB, 0, stream>>>(deg, N);

    const int aggN4   = (N * DIM) / 4;
    const int scatBlk = (E * 32 + TB - 1) / TB;
    const int gemmBlk = (N + MROWS - 1) / MROWS;

    for (int l = 0; l < L; ++l) {
        const float* Xl = (l == 0) ? node_emb : out;   // in-place after layer 0
        gnn_zero_f4<<<(aggN4 + TB - 1) / TB, TB, 0, stream>>>((float4*)agg, aggN4);
        gnn_scatter<<<scatBlk, TB, 0, stream>>>(Xl, src, dst, agg, E);
        gnn_gemm_elu<<<gemmBlk, TB, 0, stream>>>(
            Xl, agg, deg,
            Ws_rel + (size_t)l * DIM * DIM,
            bs_rel + (size_t)l * DIM,
            Ws_root + (size_t)l * DIM * DIM,
            out, N);
    }
}
